// LSTMModel_18648747999610
// MI455X (gfx1250) — matrix operand order repaired and checked by reading, not compiled, not run on hardware
//
#include <hip/hip_runtime.h>
#include <hip/hip_bf16.h>

typedef __attribute__((ext_vector_type(16))) _Float16 v16h;
typedef __attribute__((ext_vector_type(8)))  _Float16 v8h;
typedef __attribute__((ext_vector_type(8)))  float    v8f;

#define BATCH 16384
#define TT    15
#define INF   63
#define HDIM  128
#define NCLS  25

__device__ __forceinline__ float sigmoidf_(float x){ return 1.f/(1.f+__expf(-x)); }

// WMMA A fragment (16x32 f16), rows = M (batch rows), from row-major buffer
// with leading dimension ld (halfs). Contiguous k order, same as load_frag_B:
// lanes 0-15: K = kbase+[0..15]
// lanes 16-31: K = kbase+[16..31]
__device__ __forceinline__ v16h load_frag_A(const _Float16* base, int ld, int kbase, int lane){
  const _Float16* p = base + (size_t)(lane & 15) * ld + kbase + ((lane >> 4) << 4);
  v8h lo = *(const v8h*)(p);
  v8h hi = *(const v8h*)(p + 8);
  v16h r;
#pragma unroll
  for (int i = 0; i < 8; ++i){ r[i] = lo[i]; r[i+8] = hi[i]; }
  return r;
}

// WMMA B fragment (32x16 f16) of W^T, where W is (N_total x K) row-major f16.
// Lane l holds column n = nbase+(l&15) of W^T == row n of W, K-run of 16
// contiguous halfs starting at kbase + 16*(l>>4): one aligned 32B load.
__device__ __forceinline__ v16h load_frag_B(const _Float16* W, int ldw, int nbase, int kbase, int lane){
  const _Float16* p = W + (size_t)(nbase + (lane & 15)) * ldw + kbase + ((lane >> 4) << 4);
  return *(const v16h*)p;
}

__global__ void cvt_f16_kernel(const float* __restrict__ in, _Float16* __restrict__ out, int n){
  int i = blockIdx.x * 256 + threadIdx.x;
  if (i < n) out[i] = (_Float16)in[i];
}

// Fused input projection (63 -> 128) + LayerNorm, output f16 in (T, B, H) layout.
__global__ void __launch_bounds__(128) proj_ln_kernel(
    const float* __restrict__ x, const float* __restrict__ W, const float* __restrict__ pb,
    const float* __restrict__ g, const float* __restrict__ be, _Float16* __restrict__ out)
{
  const int lane = threadIdx.x & 31;
  const int wv   = threadIdx.x >> 5;
  const size_t row = (size_t)blockIdx.x * 4 + wv;   // row in [0, B*T)
  const int b = (int)(row / TT);
  const int t = (int)(row % TT);

  __shared__ __align__(16) float xr[4][64];
  const float* xp = x + row * INF;
  xr[wv][lane] = xp[lane];
  if (lane < 31) xr[wv][lane + 32] = xp[lane + 32];
  if (lane == 31) xr[wv][63] = 0.f;
  __syncthreads();

  float y[4];
#pragma unroll
  for (int i = 0; i < 4; ++i){
    int j = lane + (i << 5);
    float a = pb[j];
    const float* wr = W + j * INF;
    for (int k = 0; k < INF; ++k) a += xr[wv][k] * wr[k];
    y[i] = a;
  }
  float s1 = 0.f, s2 = 0.f;
#pragma unroll
  for (int i = 0; i < 4; ++i){ s1 += y[i]; s2 += y[i] * y[i]; }
#pragma unroll
  for (int o = 16; o > 0; o >>= 1){ s1 += __shfl_xor(s1, o, 32); s2 += __shfl_xor(s2, o, 32); }
  float mean = s1 * (1.f/128.f);
  float var  = s2 * (1.f/128.f) - mean * mean;
  float inv  = rsqrtf(var + 1e-5f);
  _Float16* op = out + ((size_t)t * BATCH + b) * HDIM;
#pragma unroll
  for (int i = 0; i < 4; ++i){
    int j = lane + (i << 5);
    op[j] = (_Float16)((y[i] - mean) * inv * g[j] + be[j]);
  }
}

// One bidirectional LSTM layer. Grid: (B/16, 2 dirs). Block: 128 (4 waves).
// Wave w computes gate w's 128 columns for 16 batch rows via WMMA; h/c in LDS.
template<int IWID>
__global__ void __launch_bounds__(128, 1) bilstm_layer(
    const _Float16* __restrict__ X,                          // (T, B, IWID) f16
    const _Float16* __restrict__ Wih_f, const _Float16* __restrict__ Whh_f, const float* __restrict__ b_f,
    const _Float16* __restrict__ Wih_b, const _Float16* __restrict__ Whh_b, const float* __restrict__ b_b,
    _Float16* __restrict__ Out)                              // (T, B, 2H) f16
{
  const int tid  = threadIdx.x;
  const int lane = tid & 31;
  const int wv   = tid >> 5;
  const int row0 = blockIdx.x << 4;
  const int dir  = blockIdx.y;
  const _Float16* Wih = dir ? Wih_b : Wih_f;
  const _Float16* Whh = dir ? Whh_b : Whh_f;
  const float*   bias = dir ? b_b  : b_f;

  __shared__ __align__(16) _Float16 hbuf[16 * 128];   // h_{t-1}, f16, 4 KB
  __shared__ __align__(16) float    cbuf[16 * 128];   // c, f32, 8 KB
  __shared__ __align__(16) float    gbuf[4][16 * 128];// gate exchange, 32 KB

  for (int idx = tid; idx < 16 * 128; idx += 128){ hbuf[idx] = (_Float16)0.f; cbuf[idx] = 0.f; }
  __syncthreads();

  const int nb0 = wv << 7;   // this wave's first gate column (gate = wv)
  float bv[8];
#pragma unroll
  for (int j = 0; j < 8; ++j) bv[j] = bias[nb0 + (j << 4) + (lane & 15)];

  for (int s = 0; s < TT; ++s){
    const int t = dir ? (TT - 1 - s) : s;
    const _Float16* xrow = X + ((size_t)t * BATCH + row0) * IWID;

    v8f acc[8];
#pragma unroll
    for (int j = 0; j < 8; ++j)
#pragma unroll
      for (int r = 0; r < 8; ++r) acc[j][r] = bv[j];

    // x_t @ Wih^T
#pragma unroll
    for (int kt = 0; kt < IWID / 32; ++kt){
      v16h a = load_frag_A(xrow, IWID, kt * 32, lane);
#pragma unroll
      for (int j = 0; j < 8; ++j){
        v16h bm = load_frag_B(Wih, IWID, nb0 + (j << 4), kt * 32, lane);
        acc[j] = __builtin_amdgcn_wmma_f32_16x16x32_f16(false, a, false, bm, (short)0, acc[j], false, false);
      }
    }
    // h_{t-1} @ Whh^T  (A fragments from LDS)
#pragma unroll
    for (int kt = 0; kt < 4; ++kt){
      v16h a = load_frag_A(hbuf, 128, kt * 32, lane);
#pragma unroll
      for (int j = 0; j < 8; ++j){
        v16h bm = load_frag_B(Whh, 128, nb0 + (j << 4), kt * 32, lane);
        acc[j] = __builtin_amdgcn_wmma_f32_16x16x32_f16(false, a, false, bm, (short)0, acc[j], false, false);
      }
    }

    // Scatter gate values to LDS (C/D layout: vgpr r -> M = r + 8*(lane>>4), N = lane&15)
#pragma unroll
    for (int j = 0; j < 8; ++j){
      int n = (j << 4) + (lane & 15);
#pragma unroll
      for (int r = 0; r < 8; ++r){
        int m = r + ((lane >> 4) << 3);
        gbuf[wv][m * 128 + n] = acc[j][r];
      }
    }
    __syncthreads();

    // Elementwise LSTM cell update: 128 threads x 16 elements = 16 rows x 128 cols
    {
      const int m  = tid >> 3;
      const int h0 = (tid & 7) << 4;
      _Float16* orow = Out + ((size_t)t * BATCH + row0 + m) * (2 * HDIM) + dir * HDIM;
#pragma unroll
      for (int k = 0; k < 16; ++k){
        int hc = h0 + k;
        int e  = m * 128 + hc;
        float iv = sigmoidf_(gbuf[0][e]);
        float fv = sigmoidf_(gbuf[1][e]);
        float gv = tanhf(gbuf[2][e]);
        float ov = sigmoidf_(gbuf[3][e]);
        float c  = fv * cbuf[e] + iv * gv;
        cbuf[e] = c;
        float h = ov * tanhf(c);
        hbuf[e] = (_Float16)h;
        orow[hc] = (_Float16)h;
      }
    }
    __syncthreads();
  }
}

// Temporal attention + classifier. One wave per batch row.
__global__ void __launch_bounds__(128) attn_cls_kernel(
    const _Float16* __restrict__ L,   // (T, B, 256) f16
    const float* __restrict__ aw,
    const float* __restrict__ W1, const float* __restrict__ b1,
    const float* __restrict__ g2, const float* __restrict__ be2,
    const float* __restrict__ W2, const float* __restrict__ b2,
    float* __restrict__ outp)
{
  const int lane = threadIdx.x & 31;
  const int wv   = threadIdx.x >> 5;
  const int b    = blockIdx.x * 4 + wv;

  __shared__ __align__(16) float ctx[4][256];
  __shared__ __align__(16) float h2s[4][128];

  float sc[TT];
#pragma unroll
  for (int t = 0; t < TT; ++t){
    const _Float16* lp = L + ((size_t)t * BATCH + b) * 256;
    float p = 0.f;
#pragma unroll
    for (int i = 0; i < 8; ++i){ int k = lane + (i << 5); p += (float)lp[k] * aw[k]; }
#pragma unroll
    for (int o = 16; o > 0; o >>= 1) p += __shfl_xor(p, o, 32);
    sc[t] = p;
  }
  float mx = sc[0];
#pragma unroll
  for (int t = 1; t < TT; ++t) mx = fmaxf(mx, sc[t]);
  float den = 0.f;
#pragma unroll
  for (int t = 0; t < TT; ++t){ sc[t] = __expf(sc[t] - mx); den += sc[t]; }
  float rden = 1.f / den;
#pragma unroll
  for (int t = 0; t < TT; ++t) sc[t] *= rden;

#pragma unroll
  for (int i = 0; i < 8; ++i){
    int k = lane + (i << 5);
    float s = 0.f;
#pragma unroll
    for (int t = 0; t < TT; ++t) s += sc[t] * (float)L[((size_t)t * BATCH + b) * 256 + k];
    ctx[wv][k] = s;
  }
  __syncthreads();

  float y[4];
#pragma unroll
  for (int i = 0; i < 4; ++i){
    int j = lane + (i << 5);
    float a = b1[j];
    const float* wr = W1 + j * 256;
    for (int k = 0; k < 256; ++k) a += ctx[wv][k] * wr[k];
    y[i] = a;
  }
  float s1 = 0.f, s2 = 0.f;
#pragma unroll
  for (int i = 0; i < 4; ++i){ s1 += y[i]; s2 += y[i] * y[i]; }
#pragma unroll
  for (int o = 16; o > 0; o >>= 1){ s1 += __shfl_xor(s1, o, 32); s2 += __shfl_xor(s2, o, 32); }
  float mean = s1 * (1.f/128.f);
  float var  = s2 * (1.f/128.f) - mean * mean;
  float inv  = rsqrtf(var + 1e-5f);
#pragma unroll
  for (int i = 0; i < 4; ++i){
    int j = lane + (i << 5);
    float v = (y[i] - mean) * inv * g2[j] + be2[j];
    v = 0.5f * v * (1.f + erff(v * 0.70710678118f));   // exact GELU
    h2s[wv][j] = v;
  }
  __syncthreads();

  if (lane < NCLS){
    float a = b2[lane];
    const float* wr = W2 + lane * 128;
    for (int k = 0; k < 128; ++k) a += h2s[wv][k] * wr[k];
    outp[(size_t)b * NCLS + lane] = a;
  }
}

extern "C" void kernel_launch(void* const* d_in, const int* in_sizes, int n_in,
                              void* d_out, int out_size, void* d_ws, size_t ws_size,
                              hipStream_t stream) {
  const float* x      = (const float*)d_in[0];
  const float* proj_W = (const float*)d_in[1];
  const float* proj_b = (const float*)d_in[2];
  const float* ln1_g  = (const float*)d_in[3];
  const float* ln1_b  = (const float*)d_in[4];
  const float* Wih0f  = (const float*)d_in[5];
  const float* Whh0f  = (const float*)d_in[6];
  const float* b0f    = (const float*)d_in[7];
  const float* Wih0b  = (const float*)d_in[8];
  const float* Whh0b  = (const float*)d_in[9];
  const float* b0b    = (const float*)d_in[10];
  const float* Wih1f  = (const float*)d_in[11];
  const float* Whh1f  = (const float*)d_in[12];
  const float* b1f    = (const float*)d_in[13];
  const float* Wih1b  = (const float*)d_in[14];
  const float* Whh1b  = (const float*)d_in[15];
  const float* b1b    = (const float*)d_in[16];
  const float* attn_w = (const float*)d_in[17];
  const float* cls_W1 = (const float*)d_in[18];
  const float* cls_b1 = (const float*)d_in[19];
  const float* ln2_g  = (const float*)d_in[20];
  const float* ln2_b  = (const float*)d_in[21];
  const float* cls_W2 = (const float*)d_in[22];
  const float* cls_b2 = (const float*)d_in[23];

  char* ws = (char*)d_ws;
  size_t off = 0;
  _Float16* ln_out = (_Float16*)(ws + off); off += (size_t)TT * BATCH * HDIM * sizeof(_Float16);
  _Float16* l0     = (_Float16*)(ws + off); off += (size_t)TT * BATCH * 256  * sizeof(_Float16);
  _Float16* l1     = (_Float16*)(ws + off); off += (size_t)TT * BATCH * 256  * sizeof(_Float16);

  auto alloc_h = [&](size_t n)->_Float16* {
    _Float16* p = (_Float16*)(ws + off); off += n * sizeof(_Float16);
    return p;
  };
  _Float16* hWih0f = alloc_h(512 * 128);
  _Float16* hWhh0f = alloc_h(512 * 128);
  _Float16* hWih0b = alloc_h(512 * 128);
  _Float16* hWhh0b = alloc_h(512 * 128);
  _Float16* hWih1f = alloc_h(512 * 256);
  _Float16* hWhh1f = alloc_h(512 * 128);
  _Float16* hWih1b = alloc_h(512 * 256);
  _Float16* hWhh1b = alloc_h(512 * 128);

  auto cvt = [&](const float* src, _Float16* dst, int n){
    cvt_f16_kernel<<<(n + 255) / 256, 256, 0, stream>>>(src, dst, n);
  };
  cvt(Wih0f, hWih0f, 512 * 128);
  cvt(Whh0f, hWhh0f, 512 * 128);
  cvt(Wih0b, hWih0b, 512 * 128);
  cvt(Whh0b, hWhh0b, 512 * 128);
  cvt(Wih1f, hWih1f, 512 * 256);
  cvt(Whh1f, hWhh1f, 512 * 128);
  cvt(Wih1b, hWih1b, 512 * 256);
  cvt(Whh1b, hWhh1b, 512 * 128);

  // Projection + LayerNorm -> (T, B, H) f16
  proj_ln_kernel<<<(BATCH * TT) / 4, 128, 0, stream>>>(x, proj_W, proj_b, ln1_g, ln1_b, ln_out);

  // BiLSTM layer 0 (input H) and layer 1 (input 2H)
  bilstm_layer<128><<<dim3(BATCH / 16, 2), 128, 0, stream>>>(
      ln_out, hWih0f, hWhh0f, b0f, hWih0b, hWhh0b, b0b, l0);
  bilstm_layer<256><<<dim3(BATCH / 16, 2), 128, 0, stream>>>(
      l0, hWih1f, hWhh1f, b1f, hWih1b, hWhh1b, b1b, l1);

  // Attention + classifier
  attn_cls_kernel<<<BATCH / 4, 128, 0, stream>>>(
      l1, attn_w, cls_W1, cls_b1, ln2_g, ln2_b, cls_W2, cls_b2, (float*)d_out);
}
